// Encoder_83631603188010
// MI455X (gfx1250) — matrix operand order repaired and checked by reading, not compiled, not run on hardware
//
#include <hip/hip_runtime.h>
#include <hip/hip_bf16.h>

typedef __attribute__((ext_vector_type(16))) __bf16 v16bf;
typedef __attribute__((ext_vector_type(8)))  float  v8f;

union ABf { unsigned u[8]; uint4 q[2]; v16bf v; };
union CF  { float f[8]; v8f v; };

__device__ __forceinline__ unsigned f2bfbits(float f) {
    unsigned u = __builtin_bit_cast(unsigned, f);
    return (u + 0x7FFFu + ((u >> 16) & 1u)) >> 16;   // round-to-nearest-even bf16
}
__device__ __forceinline__ float sigm(float x) { return 1.f / (1.f + __expf(-x)); }
__device__ __forceinline__ float tanh_fast(float x) {
    x = fminf(fmaxf(x, -15.f), 15.f);
    float e2 = __expf(2.f * x);
    return (e2 - 1.f) / (e2 + 1.f);
}

// ---------------- plain f32 -> bf16 (Whh, K=256 already 32-aligned) ----------------
__global__ void k_conv_bf16(const float* __restrict__ src,
                            unsigned short* __restrict__ dst, int n) {
    int i = blockIdx.x * blockDim.x + threadIdx.x;
    if (i < n) dst[i] = (unsigned short)f2bfbits(src[i]);
}

// ---------------- f32 [rows][K] -> bf16 [rows][Kp], zero padded ----------------
__global__ void k_conv_pad(const float* __restrict__ src, unsigned short* __restrict__ dst,
                           int rows, int K, int Kp) {
    int i = blockIdx.x * blockDim.x + threadIdx.x;
    if (i >= rows * Kp) return;
    int col = i % Kp, row = i / Kp;
    dst[i] = (col < K) ? (unsigned short)f2bfbits(src[(long long)row * K + col]) : 0;
}

// ---------------- context embedding + mask -> bf16, time-major [T*B][320] ----------------
__global__ void k_ctx_aug(const int* __restrict__ ids, const int* __restrict__ st,
                          const int* __restrict__ en, const float* __restrict__ emb,
                          unsigned short* __restrict__ aug) {
    int i = blockIdx.x * blockDim.x + threadIdx.x;     // 12800*320
    if (i >= 12800 * 320) return;
    int col = i % 320; int rem = i / 320;
    int b = rem % 32;  int t = rem / 32;
    float v = 0.f;
    if (col < 300)       v = emb[(long long)ids[b * 400 + t] * 300 + col];
    else if (col == 300) v = (st[b] <= t && en[b] >= t) ? 1.0f : 0.0f;
    aug[i] = (unsigned short)f2bfbits(v);
}

// ---------------- answer input -> bf16, time-major [TA*B][832] ----------------
__global__ void k_ans_in(const int* __restrict__ ans_ids, const int* __restrict__ st,
                         const int* __restrict__ en, const float* __restrict__ emb,
                         const float* __restrict__ ctx_out, unsigned short* __restrict__ ansin) {
    int i = blockIdx.x * blockDim.x + threadIdx.x;     // 1280*832
    if (i >= 1280 * 832) return;
    int col = i % 832; int rem = i / 832;
    int b = rem % 32;  int j = rem / 32;
    float v = 0.f;
    if (col < 512) {
        int idx = st[b] + j;
        if (idx <= en[b]) {
            int idxc = idx < 0 ? 0 : (idx > 399 ? 399 : idx);
            v = ctx_out[((long long)b * 400 + idxc) * 512 + col];
        }
    } else if (col < 812) {
        v = emb[(long long)ans_ids[b * 40 + j] * 300 + (col - 512)];
    }
    ansin[i] = (unsigned short)f2bfbits(v);
}

// ---------------- input projection: Y[M][1024] = Xbf[M][Kp] @ Wbf[1024][Kp]^T + bias ------
// each wave owns 1 M-tile x 4 N-tiles; A fragment reused across 4 WMMAs per K-step.
// All operands pre-padded bf16 -> pure b128 loads, no guards, no conversion.
__global__ void __launch_bounds__(256) k_proj_wmma(
    const unsigned short* __restrict__ Xbf, const unsigned short* __restrict__ Wbf,
    const float* __restrict__ bias, float* __restrict__ Y, int Kp) {
    const int lane = threadIdx.x & 31;
    const int wave = threadIdx.x >> 5;
    const int l15 = lane & 15, half = lane >> 4;
    int wid = blockIdx.x * 8 + wave;
    int tM  = wid >> 4;            // M/16 tiles
    int tN4 = wid & 15;            // 16 groups of 4 N-tiles = 64 N-tiles
    const unsigned short* Xrow = Xbf + (long long)(tM * 16 + l15) * Kp;
    CF acc[4];
#pragma unroll
    for (int g = 0; g < 4; ++g)
#pragma unroll
        for (int r = 0; r < 8; ++r) acc[g].f[r] = 0.f;
    int nkk = Kp >> 5;
    for (int kk = 0; kk < nkk; ++kk) {
        ABf a;
        a.q[0] = *reinterpret_cast<const uint4*>(Xrow + kk * 32 + half * 8);        // K 0-7 | 8-15
        a.q[1] = *reinterpret_cast<const uint4*>(Xrow + kk * 32 + 16 + half * 8);   // K 16-23 | 24-31
#pragma unroll
        for (int g = 0; g < 4; ++g) {
            int n = (tN4 * 4 + g) * 16 + l15;
            const unsigned short* Wrow = Wbf + (long long)n * Kp + kk * 32 + half * 8;
            ABf bm;
            bm.q[0] = *reinterpret_cast<const uint4*>(Wrow);
            bm.q[1] = *reinterpret_cast<const uint4*>(Wrow + 16);
            acc[g].v = __builtin_amdgcn_wmma_f32_16x16x32_bf16(
                false, a.v, false, bm.v, (short)0, acc[g].v, false, false);
        }
    }
#pragma unroll
    for (int g = 0; g < 4; ++g) {
        int n = (tN4 * 4 + g) * 16 + l15;
        float bb = bias[n];
#pragma unroll
        for (int r = 0; r < 8; ++r) {
            int m = tM * 16 + half * 8 + r;
            Y[(long long)m * 1024 + n] = acc[g].f[r] + bb;
        }
    }
}

// ---------------- persistent BiLSTM recurrence: one workgroup per direction ----------------
__global__ void __launch_bounds__(256) k_lstm_rec(
    const float* __restrict__ xg_f, const float* __restrict__ xg_b,
    const unsigned short* __restrict__ whh_f, const unsigned short* __restrict__ whh_b,
    float* __restrict__ out, int T) {
    const int dir = blockIdx.x;
    const float* xg = dir ? xg_b : xg_f;
    const unsigned short* W = dir ? whh_b : whh_f;
    __shared__ __align__(16) unsigned short hbf[32 * 256];   // 16 KB
    const int lane = threadIdx.x & 31;
    const int wave = threadIdx.x >> 5;
    const int l15 = lane & 15, half = lane >> 4;

    for (int i = threadIdx.x; i < 32 * 256; i += 256) hbf[i] = 0;
    float c[4][8];
#pragma unroll
    for (int g = 0; g < 4; ++g)
#pragma unroll
        for (int r = 0; r < 8; ++r) c[g][r] = 0.f;
    __syncthreads();

    for (int s = 0; s < T; ++s) {
        int t = dir ? (T - 1 - s) : s;
        int tn = dir ? (t - 1 < 0 ? 0 : t - 1) : (t + 1 >= T ? T - 1 : t + 1);
        __builtin_prefetch(xg + (long long)tn * 32 * 1024, 0, 1);  // global_prefetch_b8

        float hnew[4][8];
#pragma unroll
        for (int gi = 0; gi < 4; ++gi) {
            int grp = wave * 4 + gi;        // 32 groups: 2 mtiles x 16 h-tiles
            int mtile = grp >> 4;
            int htile = grp & 15;
            CF acc[4];
#pragma unroll
            for (int g = 0; g < 4; ++g)
#pragma unroll
                for (int r = 0; r < 8; ++r) acc[g].f[r] = 0.f;
            const unsigned short* hrow = &hbf[(mtile * 16 + l15) * 256];
#pragma unroll
            for (int kk = 0; kk < 8; ++kk) {
                ABf a;
                a.q[0] = *reinterpret_cast<const uint4*>(hrow + kk * 32 + half * 8);
                a.q[1] = *reinterpret_cast<const uint4*>(hrow + kk * 32 + 16 + half * 8);
#pragma unroll
                for (int g = 0; g < 4; ++g) {   // i,f,g,o tiles share this A fragment
                    int n = g * 256 + htile * 16 + l15;
                    const unsigned short* Wrow = W + n * 256 + kk * 32 + half * 8;
                    ABf bm;
                    bm.q[0] = *reinterpret_cast<const uint4*>(Wrow);
                    bm.q[1] = *reinterpret_cast<const uint4*>(Wrow + 16);
                    acc[g].v = __builtin_amdgcn_wmma_f32_16x16x32_bf16(
                        false, a.v, false, bm.v, (short)0, acc[g].v, false, false);
                }
            }
            int h_idx = htile * 16 + l15;
            const float* xgt = xg + (long long)t * 32 * 1024;
#pragma unroll
            for (int r = 0; r < 8; ++r) {
                int m = mtile * 16 + half * 8 + r;
                float gi_ = acc[0].f[r] + xgt[m * 1024 +       h_idx];
                float gf_ = acc[1].f[r] + xgt[m * 1024 + 256 + h_idx];
                float gg_ = acc[2].f[r] + xgt[m * 1024 + 512 + h_idx];
                float go_ = acc[3].f[r] + xgt[m * 1024 + 768 + h_idx];
                float cc = sigm(gf_) * c[gi][r] + sigm(gi_) * tanh_fast(gg_);
                c[gi][r] = cc;
                hnew[gi][r] = sigm(go_) * tanh_fast(cc);
            }
        }
        __syncthreads();   // all LDS h reads done before overwrite
#pragma unroll
        for (int gi = 0; gi < 4; ++gi) {
            int grp = wave * 4 + gi;
            int mtile = grp >> 4, htile = grp & 15;
            int h_idx = htile * 16 + l15;
#pragma unroll
            for (int r = 0; r < 8; ++r) {
                int m = mtile * 16 + half * 8 + r;
                hbf[m * 256 + h_idx] = (unsigned short)f2bfbits(hnew[gi][r]);
                out[(long long)m * T * 512 + (long long)t * 512 + dir * 256 + h_idx] = hnew[gi][r];
            }
        }
        __syncthreads();   // new h visible before next step
    }
}

// ---------------- tail: means, FC1 + residual, FC2 + tanh ----------------
__global__ void k_mean(const float* __restrict__ ans_out, const float* __restrict__ ctx_out,
                       float* __restrict__ ared, float* __restrict__ cred) {
    int i = blockIdx.x * blockDim.x + threadIdx.x;
    if (i >= 32 * 512) return;
    int b = i >> 9, jc = i & 511;
    float sa = 0.f, sc = 0.f;
    for (int t = 0; t < 40; ++t)  sa += ans_out[((long long)b * 40 + t) * 512 + jc];
    for (int t = 0; t < 400; ++t) sc += ctx_out[((long long)b * 400 + t) * 512 + jc];
    ared[i] = sa * (1.f / 40.f);
    cred[i] = sc * (1.f / 400.f);
}

__global__ void k_fusion(const float* __restrict__ ared, const float* __restrict__ cred,
                         const float* __restrict__ fc1W, const float* __restrict__ fc1b,
                         float* __restrict__ fus) {
    int i = blockIdx.x * blockDim.x + threadIdx.x;
    if (i >= 32 * 512) return;
    int b = i >> 9, n = i & 511;
    float s = fc1b[n] + cred[i];
    const float* a = ared + b * 512;
    const float* w = fc1W + n * 512;
    for (int j = 0; j < 512; ++j) s += a[j] * w[j];
    fus[i] = s;
}

__global__ void k_hidden(const float* __restrict__ fus, const float* __restrict__ fc2W,
                         const float* __restrict__ fc2b, float* __restrict__ hid) {
    int i = blockIdx.x * blockDim.x + threadIdx.x;
    if (i >= 32 * 512) return;
    int b = i >> 9, d = i & 511;
    float s = fc2b[d];
    const float* f = fus + b * 512;
    const float* w = fc2W + d * 512;
    for (int n = 0; n < 512; ++n) s += f[n] * w[n];
    hid[i] = tanh_fast(s);
}

extern "C" void kernel_launch(void* const* d_in, const int* in_sizes, int n_in,
                              void* d_out, int out_size, void* d_ws, size_t ws_size,
                              hipStream_t stream) {
    (void)in_sizes; (void)n_in; (void)out_size; (void)ws_size;
    const int*   ctx_ids = (const int*)d_in[0];
    const int*   ans_ids = (const int*)d_in[1];
    const int*   st      = (const int*)d_in[2];
    const int*   en      = (const int*)d_in[3];
    const float* emb     = (const float*)d_in[4];
    const float* ctxWihF = (const float*)d_in[5];
    const float* ctxWhhF = (const float*)d_in[6];
    const float* ctxbF   = (const float*)d_in[7];
    const float* ctxWihB = (const float*)d_in[8];
    const float* ctxWhhB = (const float*)d_in[9];
    const float* ctxbB   = (const float*)d_in[10];
    const float* ansWihF = (const float*)d_in[11];
    const float* ansWhhF = (const float*)d_in[12];
    const float* ansbF   = (const float*)d_in[13];
    const float* ansWihB = (const float*)d_in[14];
    const float* ansWhhB = (const float*)d_in[15];
    const float* ansbB   = (const float*)d_in[16];
    const float* fc1W    = (const float*)d_in[17];
    const float* fc1b    = (const float*)d_in[18];
    const float* fc2W    = (const float*)d_in[19];
    const float* fc2b    = (const float*)d_in[20];

    float* ctx_out = (float*)d_out;                              // [32][400][512]
    float* hidden  = (float*)d_out + (size_t)32 * 400 * 512;     // [32][512]

    const int KpC = 320, KpA = 832;
    char* w = (char*)d_ws;
    const int nW = 1024 * 256;
    unsigned short* whh = (unsigned short*)w;                    // 4 x Whh bf16
    size_t off = (size_t)4 * nW * 2;
    unsigned short* wihCF = (unsigned short*)(w + off); off += 1024ull * KpC * 2;
    unsigned short* wihCB = (unsigned short*)(w + off); off += 1024ull * KpC * 2;
    unsigned short* wihAF = (unsigned short*)(w + off); off += 1024ull * KpA * 2;
    unsigned short* wihAB = (unsigned short*)(w + off); off += 1024ull * KpA * 2;
    unsigned short* aug   = (unsigned short*)(w + off); off += 12800ull * KpC * 2;
    unsigned short* ansin = (unsigned short*)(w + off); off += 1280ull * KpA * 2;
    float* xgCF = (float*)(w + off); off += 400ull * 32 * 1024 * 4;
    float* xgCB = (float*)(w + off); off += 400ull * 32 * 1024 * 4;
    float* xgAF = (float*)(w + off); off += 40ull  * 32 * 1024 * 4;
    float* xgAB = (float*)(w + off); off += 40ull  * 32 * 1024 * 4;
    float* ansO = (float*)(w + off); off += 32ull  * 40 * 512 * 4;
    float* ared = (float*)(w + off); off += 32ull  * 512 * 4;
    float* cred = (float*)(w + off); off += 32ull  * 512 * 4;
    float* fus  = (float*)(w + off); off += 32ull  * 512 * 4;

    // weight conversions (bf16, padded where needed)
    k_conv_bf16<<<(nW + 255) / 256, 256, 0, stream>>>(ctxWhhF, whh + 0 * nW, nW);
    k_conv_bf16<<<(nW + 255) / 256, 256, 0, stream>>>(ctxWhhB, whh + 1 * nW, nW);
    k_conv_bf16<<<(nW + 255) / 256, 256, 0, stream>>>(ansWhhF, whh + 2 * nW, nW);
    k_conv_bf16<<<(nW + 255) / 256, 256, 0, stream>>>(ansWhhB, whh + 3 * nW, nW);
    { int n = 1024 * KpC;
      k_conv_pad<<<(n + 255) / 256, 256, 0, stream>>>(ctxWihF, wihCF, 1024, 301, KpC);
      k_conv_pad<<<(n + 255) / 256, 256, 0, stream>>>(ctxWihB, wihCB, 1024, 301, KpC); }
    { int n = 1024 * KpA;
      k_conv_pad<<<(n + 255) / 256, 256, 0, stream>>>(ansWihF, wihAF, 1024, 812, KpA);
      k_conv_pad<<<(n + 255) / 256, 256, 0, stream>>>(ansWihB, wihAB, 1024, 812, KpA); }

    { int n = 12800 * KpC;
      k_ctx_aug<<<(n + 255) / 256, 256, 0, stream>>>(ctx_ids, st, en, emb, aug); }

    { int blocks = 800 * 16 / 8;   // 1600: waves = Mtiles(800) x N-groups(16)
      k_proj_wmma<<<blocks, 256, 0, stream>>>(aug, wihCF, ctxbF, xgCF, KpC);
      k_proj_wmma<<<blocks, 256, 0, stream>>>(aug, wihCB, ctxbB, xgCB, KpC); }

    k_lstm_rec<<<2, 256, 0, stream>>>(xgCF, xgCB, whh + 0 * nW, whh + 1 * nW, ctx_out, 400);

    { int n = 1280 * KpA;
      k_ans_in<<<(n + 255) / 256, 256, 0, stream>>>(ans_ids, st, en, emb, ctx_out, ansin); }

    { int blocks = 80 * 16 / 8;    // 160
      k_proj_wmma<<<blocks, 256, 0, stream>>>(ansin, wihAF, ansbF, xgAF, KpA);
      k_proj_wmma<<<blocks, 256, 0, stream>>>(ansin, wihAB, ansbB, xgAB, KpA); }

    k_lstm_rec<<<2, 256, 0, stream>>>(xgAF, xgAB, whh + 2 * nW, whh + 3 * nW, ansO, 40);

    { int n = 32 * 512;
      k_mean  <<<(n + 255) / 256, 256, 0, stream>>>(ansO, ctx_out, ared, cred);
      k_fusion<<<(n + 255) / 256, 256, 0, stream>>>(ared, cred, fc1W, fc1b, fus);
      k_hidden<<<(n + 255) / 256, 256, 0, stream>>>(fus, fc2W, fc2b, hidden); }
}
